// InteractionLayer_52621939311306
// MI455X (gfx1250) — matrix operand order repaired and checked by reading, not compiled, not run on hardware
//
#include <hip/hip_runtime.h>
#include <hip/hip_bf16.h>

typedef __attribute__((ext_vector_type(16))) _Float16 v16h;
typedef __attribute__((ext_vector_type(8)))  float    v8f;

#define NROWS 131072
#define IDIM  256
#define ODIM  256
#define LDIM  128   // K dimension of both GEMMs

// LDS row stride for the staged x tile: 260 floats.
//  - 260*4 = 1040 B, 16B-aligned -> float4 LDS stores stay aligned
//  - 260 mod 64 = 4 -> bank(row,col) = (4*row + col) % 64; 16 consecutive
//    rows at a fixed column hit 16 distinct banks (conflict-free gathers)
#define XS_STRIDE 260

// ---------------------------------------------------------------------------
// Kernel 1: reciprocal column norms of W0 and W1 (columns have 128 entries).
// One thread per column; loads are coalesced across threads.
// ---------------------------------------------------------------------------
__global__ void il_colnorm_kernel(const float* __restrict__ W0,
                                  const float* __restrict__ W1,
                                  float* __restrict__ rnorms) {
  int c = blockIdx.x * blockDim.x + threadIdx.x;     // 0..511
  if (c >= 2 * ODIM) return;
  const float* W = (c < ODIM) ? W0 : W1;
  int col = c & (ODIM - 1);
  float s = 0.f;
  for (int k = 0; k < LDIM; ++k) {
    float v = W[k * ODIM + col];
    s += v * v;
  }
  rnorms[c] = rsqrtf(s);
}

// ---------------------------------------------------------------------------
// Kernel 2: normalize weights, convert to f16, pre-swizzle into the wave32
// WMMA B-fragment layout for V_WMMA_F32_16X16X32_F16:
//   flat half index t = (((mat*16 + ntile)*4 + ks)*32 + lane)*16 + h
//   maps to element (k, n) with n = ntile*16 + (lane&15),
//                              k = ks*32 + (h&7) + 16*(h>>3) + 8*(lane>>4)
// so in the GEMM each lane loads its 16-half fragment contiguously (32B).
// ---------------------------------------------------------------------------
__global__ void il_pack_kernel(const float* __restrict__ W0,
                               const float* __restrict__ W1,
                               const float* __restrict__ rnorms,
                               _Float16* __restrict__ Bp) {
  int t = blockIdx.x * blockDim.x + threadIdx.x;     // 0..65535
  int h     = t & 15;
  int lane  = (t >> 4) & 31;
  int ks    = (t >> 9) & 3;
  int ntile = (t >> 11) & 15;
  int mat   = (t >> 15) & 1;
  int n = ntile * 16 + (lane & 15);
  int k = ks * 32 + (h & 7) + 16 * (h >> 3) + 8 * (lane >> 4);
  const float* W = mat ? W1 : W0;
  float v = W[k * ODIM + n] * rnorms[mat * ODIM + n];
  Bp[t] = (_Float16)v;
}

// ---------------------------------------------------------------------------
// Kernel 3: main interaction kernel.
// Block = 256 threads = 8 wave32s, covering 128 rows of x.
// Phase 1: cooperatively stage the block's 128 x-rows (f32) into padded LDS
//          with one fully-coalesced float4 sweep (each HBM byte read once).
// Phase 2: each wave owns a 16-row M-slice; builds both gathered A fragments
//          (f32->f16, ISA 16-bit A 16x32 layout) from LDS, then runs 16
//          N-tiles x 4 K-steps x 2 GEMMs of v_wmma_f32_16x16x32_f16.
// Epilogue: out = x[:, right_idx] (from LDS) + s0*s1, per the C layout.
// LDS: 128*260*4 = 133120 B + 2 KB indices -> 2 workgroups per 320KB WGP.
// ---------------------------------------------------------------------------
__global__ __launch_bounds__(256)
void il_interaction_kernel(const float* __restrict__ x,
                           const _Float16* __restrict__ Bp,
                           const int* __restrict__ right_idx,
                           const int* __restrict__ left0,
                           const int* __restrict__ left1,
                           float* __restrict__ out) {
  __shared__ float xs[128 * XS_STRIDE];
  __shared__ int   sl0[LDIM];
  __shared__ int   sl1[LDIM];
  __shared__ int   sright[ODIM];

  const int lane = threadIdx.x & 31;
  const int wave = threadIdx.x >> 5;                  // 0..7
  const int lmod = lane & 15;
  const int lhi  = lane >> 4;                         // 0 or 1
  const int m_base = blockIdx.x * 128 + wave * 16;    // 16 rows per wave

  // ---- Phase 1: coalesced stage of 128 rows x 256 cols f32 into LDS.
  {
    const float4* xg = (const float4*)(x + (size_t)blockIdx.x * 128 * IDIM);
    for (int i = threadIdx.x; i < 128 * (IDIM / 4); i += 256) {
      int row = i >> 6;             // IDIM/4 = 64 float4 per row
      int c4  = i & 63;
      float4 v = xg[row * 64 + c4];
      *(float4*)&xs[row * XS_STRIDE + c4 * 4] = v;
    }
    if (threadIdx.x < LDIM) {
      sl0[threadIdx.x] = left0[threadIdx.x];
      sl1[threadIdx.x] = left1[threadIdx.x];
    }
    sright[threadIdx.x] = right_idx[threadIdx.x];     // 256 entries
  }
  __syncthreads();

  // ---- Phase 2: build A fragments for both gathers from LDS.
  // ISA 16-bit A 16x32 layout: lane&15 selects M-row; half h of k-step ks
  // holds k = ks*32 + (h&7) + 16*(h>>3) + 8*(lane>>4)
  const float* xrow = &xs[(wave * 16 + lmod) * XS_STRIDE];
  v16h A0[4], A1[4];
#pragma unroll
  for (int ks = 0; ks < 4; ++ks) {
    v16h a0, a1;
#pragma unroll
    for (int h = 0; h < 16; ++h) {
      int k = ks * 32 + (h & 7) + ((h >> 3) << 4) + (lhi << 3);
      a0[h] = (_Float16)xrow[sl0[k]];
      a1[h] = (_Float16)xrow[sl1[k]];
    }
    A0[ks] = a0;
    A1[ks] = a1;
  }

  const v16h* Bv = (const v16h*)Bp;  // fragment index = ((mat*16+nt)*4+ks)*32 + lane

  // ---- 16 N-tiles of 16x16 outputs
  for (int nt = 0; nt < 16; ++nt) {
    v8f c0 = {};
    v8f c1 = {};
#pragma unroll
    for (int ks = 0; ks < 4; ++ks) {
      v16h b0 = Bv[((0 * 16 + nt) * 4 + ks) * 32 + lane];
      v16h b1 = Bv[((1 * 16 + nt) * 4 + ks) * 32 + lane];
      c0 = __builtin_amdgcn_wmma_f32_16x16x32_f16(
          false, A0[ks], false, b0, (short)0, c0, false, false);
      c1 = __builtin_amdgcn_wmma_f32_16x16x32_f16(
          false, A1[ks], false, b1, (short)0, c1, false, false);
    }

    // ---- Epilogue: out = x[:, right_idx] + s0*s1
    // C/D layout: lanes 0-15 hold M=v (VGPR v), lanes 16-31 hold M=v+8; N=lane&15.
    int n    = nt * 16 + lmod;
    int rcol = sright[n];
#pragma unroll
    for (int v = 0; v < 8; ++v) {
      int mrow_l = wave * 16 + v + 8 * lhi;           // row within LDS tile
      float g = xs[mrow_l * XS_STRIDE + rcol];
      out[(m_base + v + 8 * lhi) * ODIM + n] = g + c0[v] * c1[v];
    }
  }
}

// ---------------------------------------------------------------------------
extern "C" void kernel_launch(void* const* d_in, const int* in_sizes, int n_in,
                              void* d_out, int out_size, void* d_ws, size_t ws_size,
                              hipStream_t stream) {
  const float* x     = (const float*)d_in[0];
  const float* W0    = (const float*)d_in[1];
  const float* W1    = (const float*)d_in[2];
  const int*   right = (const int*)d_in[3];
  const int*   l0    = (const int*)d_in[4];
  const int*   l1    = (const int*)d_in[5];
  float* out = (float*)d_out;

  // Workspace layout: [0,2048) reciprocal norms (512 f32),
  //                   [2048, 2048+131072) packed f16 B fragments (2*128*256).
  float*    rnorms = (float*)d_ws;
  _Float16* Bp     = (_Float16*)((char*)d_ws + 2048);

  il_colnorm_kernel<<<2, 256, 0, stream>>>(W0, W1, rnorms);
  il_pack_kernel<<<256, 256, 0, stream>>>(W0, W1, rnorms, Bp);
  il_interaction_kernel<<<NROWS / 128, 256, 0, stream>>>(x, Bp, right, l0, l1, out);
}
